// NormMixAttention_13176959664464
// MI455X (gfx1250) — matrix operand order repaired and checked by reading, not compiled, not run on hardware
//
#include <hip/hip_runtime.h>

// ---------------- problem constants ----------------
constexpr int kN  = 4096;          // tokens
constexpr int kB  = 4;             // batch
constexpr int kE  = 1024;          // embed
constexpr int kH  = 16;            // heads
constexpr int kC  = 64;            // chunk
constexpr int kD  = 512;           // inner dim (E/2)
constexpr int kDH = 32;            // per-head dim
constexpr int kL  = 64;            // chunks
constexpr int kM  = kN * kB;       // 16384 rows

typedef __attribute__((ext_vector_type(16))) __bf16         v16bf;
typedef __attribute__((ext_vector_type(8)))  float          v8f;
typedef __attribute__((ext_vector_type(8)))  unsigned short v8u;

union V16 { v8u h[2]; v16bf b; };

__device__ inline v8f v8f_zero() {
  v8f z = {0.f,0.f,0.f,0.f,0.f,0.f,0.f,0.f};
  return z;
}
__device__ inline unsigned short f2bf(float f) {
  unsigned int u = __float_as_uint(f);
  u += 0x7FFFu + ((u >> 16) & 1u);              // round-to-nearest-even
  return (unsigned short)(u >> 16);
}
__device__ inline float geluf(float x) {       // jax.nn.gelu (tanh approx)
  float x3 = x * x * x;
  return 0.5f * x * (1.0f + tanhf(0.7978845608028654f * (x + 0.044715f * x3)));
}
__device__ inline float sigmf(float x) { return 1.0f / (1.0f + __expf(-x)); }

__device__ inline v8f wmma_bf16(const V16& a, const V16& b, v8f c) {
  return __builtin_amdgcn_wmma_f32_16x16x32_bf16(false, a.b, false, b.b,
                                                 (short)0, c, false, false);
}

// CDNA5 async VGPR-free global->LDS copy (ASYNCcnt-tracked), ISA 15.18.3.
// Low 32 bits of a flat pointer into __shared__ are the LDS byte offset.
__device__ inline void async_ld128(unsigned short* lds, const unsigned short* g) {
  unsigned l = (unsigned)(size_t)lds;
  asm volatile("global_load_async_to_lds_b128 %0, %1, off"
               :: "v"(l), "v"(g) : "memory");
}
__device__ inline void wait_async0() {
  asm volatile("s_wait_asynccnt 0x0" ::: "memory");
}

// ---------------- f32 -> bf16 conversion ----------------
__global__ void k_f2bf(const float* __restrict__ in, unsigned short* __restrict__ out, int n) {
  int i = blockIdx.x * blockDim.x + threadIdx.x;
  if (i < n) out[i] = f2bf(in[i]);
}

// ---------------- bf16 GEMM:  C[m,n] = epilogue( sum_k A[m,k]*W[n,k] + bias[n] ) ----------
// 128x128x32 block tile, 8 waves of 32x64, double-buffered async-to-LDS staging.
// ACT: 0 none, 1 gelu.  MODE: 0 store bf16, 1 store f32, 2 C = 0.5*(C + val) f32.
template<int ACT, int MODE>
__global__ __launch_bounds__(256)
void k_gemm(const unsigned short* __restrict__ A,   // M x K bf16, row-major
            const unsigned short* __restrict__ W,   // N x K bf16, row-major
            const float* __restrict__ bias,
            void* __restrict__ Cout,
            int M, int N, int K)
{
  __shared__ unsigned short As[2][128 * 32];
  __shared__ unsigned short Ws[2][128 * 32];
  const int tid  = threadIdx.x;
  const int lane = tid & 31, wid = tid >> 5;
  const int wm = wid & 3, wn = wid >> 2;           // 4 x 2 wave grid, each 32x64
  const int m0 = blockIdx.y * 128;
  const int n0 = blockIdx.x * 128;
  const int l15 = lane & 15;
  const int hi  = lane >> 4;
  const int abase = 8 * hi;                        // A K-chunk base (halves)
  const int bko   = 8 * hi;                        // B K base (halves)

  v8f acc[2][4];
  for (int i = 0; i < 2; i++) for (int j = 0; j < 4; j++) acc[i][j] = v8f_zero();

  const int srow = tid >> 1, skoff = (tid & 1) * 16;   // 128 rows x 32 halves per tile

  auto stage = [&](int buf, int k0) {
    size_t ga = (size_t)(m0 + srow) * K + k0 + skoff;
    async_ld128(&As[buf][srow * 32 + skoff],     A + ga);
    async_ld128(&As[buf][srow * 32 + skoff + 8], A + ga + 8);
    size_t gw = (size_t)(n0 + srow) * K + k0 + skoff;
    async_ld128(&Ws[buf][srow * 32 + skoff],     W + gw);
    async_ld128(&Ws[buf][srow * 32 + skoff + 8], W + gw + 8);
  };

  stage(0, 0);
  wait_async0();
  __syncthreads();

  const int nk = K >> 5;
  for (int it = 0; it < nk; ++it) {
    const int buf = it & 1;
    if (it + 1 < nk) stage(buf ^ 1, (it + 1) << 5);   // overlap next tile with compute

    V16 av[2], bv[4];
    for (int mi = 0; mi < 2; mi++) {
      int r = wm * 32 + mi * 16 + l15;
      av[mi].h[0] = *(const v8u*)(&As[buf][r * 32 + abase]);
      av[mi].h[1] = *(const v8u*)(&As[buf][r * 32 + abase + 16]);
    }
    for (int ni = 0; ni < 4; ni++) {
      int r = wn * 64 + ni * 16 + l15;
      bv[ni].h[0] = *(const v8u*)(&Ws[buf][r * 32 + bko]);
      bv[ni].h[1] = *(const v8u*)(&Ws[buf][r * 32 + bko + 16]);
    }
    for (int mi = 0; mi < 2; mi++)
      for (int ni = 0; ni < 4; ni++)
        acc[mi][ni] = wmma_bf16(av[mi], bv[ni], acc[mi][ni]);

    wait_async0();          // async writes into buf^1 complete
    __syncthreads();        // all waves done reading buf before it is restaged
  }

  const int rhi = 8 * hi;
  for (int mi = 0; mi < 2; mi++) {
    for (int ni = 0; ni < 4; ni++) {
      int col = n0 + wn * 64 + ni * 16 + l15;
      float bsv = bias[col];
      for (int j = 0; j < 8; j++) {
        int row = m0 + wm * 32 + mi * 16 + j + rhi;
        float v = acc[mi][ni][j] + bsv;
        if (ACT == 1) v = geluf(v);
        if (MODE == 0) {
          ((unsigned short*)Cout)[(size_t)row * N + col] = f2bf(v);
        } else if (MODE == 1) {
          ((float*)Cout)[(size_t)row * N + col] = v;
        } else {
          float* p = (float*)Cout + (size_t)row * N + col;
          *p = 0.5f * (*p + v);
        }
      }
    }
  }
}

// ---------------- chunked attention (one block per (b,h), serial over L chunks) -------
// LINEAR=true : causal masked S, plus kv-state recurrence (inter-chunk term).
// LINEAR=false: relu(S)*mask, intra only.
template<bool LINEAR>
__global__ __launch_bounds__(256)
void k_attn(const unsigned short* __restrict__ Qm,
            const unsigned short* __restrict__ Km,
            const unsigned short* __restrict__ Vm,
            float* __restrict__ O)
{
  __shared__ unsigned short qs[kC * kDH];      // [c][d]
  __shared__ unsigned short ks[kC * kDH];      // [c][d]
  __shared__ unsigned short vT[kDH * kC];      // [d][c]
  __shared__ unsigned short kT[kDH * kC];      // [d][c]  (linear only)
  __shared__ unsigned short Ss[kC * kC];       // [c][c'] bf16
  __shared__ unsigned short kvT[kDH * kDH];    // [e][d]  kv_prev transposed, bf16

  const int bid = blockIdx.x;
  const int h = bid & (kH - 1);
  const int b = bid >> 4;
  const int tid  = threadIdx.x;
  const int lane = tid & 31, wid = tid >> 5;
  const int l15 = lane & 15;
  const int hi  = lane >> 4;
  const int abase = 8 * hi, bko = 8 * hi, rhi = 8 * hi;

  const int c    = tid >> 2;            // 0..63  (staging row)
  const int koff = (tid & 3) * 8;       // 0..24  (staging k offset)

  v8f kvacc = v8f_zero();               // persistent kv state (waves 0..3)
  const int di = wid & 1, ej = wid >> 1;

  for (int l = 0; l < kL; ++l) {
    // publish kv_prev (state before this chunk's update) as bf16, transposed
    if (LINEAR && wid < 4) {
      for (int j = 0; j < 8; j++) {
        int d = di * 16 + j + rhi;
        int e = ej * 16 + l15;
        kvT[e * kDH + d] = f2bf(kvacc[j]);
      }
    }
    // stage q, k, v chunk (and transposed copies)
    {
      size_t grow = ((size_t)(l * kC + c) * kB + b) * kD + h * kDH + koff;
      v8u qv = *(const v8u*)(Qm + grow);
      v8u kv = *(const v8u*)(Km + grow);
      v8u vv = *(const v8u*)(Vm + grow);
      *(v8u*)(qs + c * kDH + koff) = qv;
      *(v8u*)(ks + c * kDH + koff) = kv;
      for (int i = 0; i < 8; i++) {
        vT[(koff + i) * kC + c] = vv[i];
        if (LINEAR) kT[(koff + i) * kC + c] = kv[i];
      }
    }
    __syncthreads();

    // S = q . k^T  (64x64), mask / relu, store bf16
    {
      const int ti  = wid & 3;
      const int tj0 = (wid >> 2) * 2;
      V16 a;
      int r = ti * 16 + l15;
      a.h[0] = *(const v8u*)(qs + r * kDH + abase);
      a.h[1] = *(const v8u*)(qs + r * kDH + abase + 16);
      for (int t = 0; t < 2; t++) {
        int tj = tj0 + t;
        V16 bb;
        int rc = tj * 16 + l15;
        bb.h[0] = *(const v8u*)(ks + rc * kDH + bko);
        bb.h[1] = *(const v8u*)(ks + rc * kDH + bko + 16);
        v8f s = wmma_bf16(a, bb, v8f_zero());
        for (int j = 0; j < 8; j++) {
          int rr = ti * 16 + j + rhi;
          int cc = tj * 16 + l15;
          float v = s[j];
          if (!LINEAR) v = fmaxf(v, 0.0f);
          if (cc > rr) v = 0.0f;                 // causal in-chunk mask
          Ss[rr * kC + cc] = f2bf(v);
        }
      }
    }
    __syncthreads();

    // o = S.v (+ q.kv_prev), and kv += k^T.v
    {
      const int ti = wid & 3;
      const int dj = wid >> 2;
      v8f oacc = v8f_zero();
      if (LINEAR) {                               // inter-chunk term
        V16 a, bb;
        int r = ti * 16 + l15;
        a.h[0] = *(const v8u*)(qs + r * kDH + abase);
        a.h[1] = *(const v8u*)(qs + r * kDH + abase + 16);
        int re = dj * 16 + l15;
        bb.h[0] = *(const v8u*)(kvT + re * kDH + bko);
        bb.h[1] = *(const v8u*)(kvT + re * kDH + bko + 16);
        oacc = wmma_bf16(a, bb, oacc);
      }
      for (int s = 0; s < 2; s++) {               // intra: K = 64 in 2 steps
        V16 a, bb;
        int r = ti * 16 + l15;
        a.h[0] = *(const v8u*)(Ss + r * kC + s * 32 + abase);
        a.h[1] = *(const v8u*)(Ss + r * kC + s * 32 + abase + 16);
        int rd = dj * 16 + l15;
        bb.h[0] = *(const v8u*)(vT + rd * kC + s * 32 + bko);
        bb.h[1] = *(const v8u*)(vT + rd * kC + s * 32 + bko + 16);
        oacc = wmma_bf16(a, bb, oacc);
      }
      for (int j = 0; j < 8; j++) {
        int cc = ti * 16 + j + rhi;
        size_t row = (size_t)(l * kC + cc) * kB + b;
        O[row * kD + h * kDH + dj * 16 + l15] = oacc[j];
      }
      if (LINEAR && wid < 4) {                    // kv state update: k^T . v
        for (int s = 0; s < 2; s++) {
          V16 a, bb;
          int rd = di * 16 + l15;
          a.h[0] = *(const v8u*)(kT + rd * kC + s * 32 + abase);
          a.h[1] = *(const v8u*)(kT + rd * kC + s * 32 + abase + 16);
          int rv = ej * 16 + l15;
          bb.h[0] = *(const v8u*)(vT + rv * kC + s * 32 + bko);
          bb.h[1] = *(const v8u*)(vT + rv * kC + s * 32 + bko + 16);
          kvacc = wmma_bf16(a, bb, kvacc);
        }
      }
    }
    __syncthreads();
  }
}

// ---------------- LayerNorm over D=512, output bf16 ----------------
__global__ __launch_bounds__(256)
void k_layernorm(const float* __restrict__ X, const float* __restrict__ g,
                 const float* __restrict__ bt, unsigned short* __restrict__ Y)
{
  __shared__ float s1[256], s2[256];
  int row = blockIdx.x, t = threadIdx.x;
  const float* x = X + (size_t)row * kD;
  float x0 = x[t], x1 = x[t + 256];
  s1[t] = x0 + x1; s2[t] = x0 * x0 + x1 * x1;
  __syncthreads();
  for (int s = 128; s > 0; s >>= 1) {
    if (t < s) { s1[t] += s1[t + s]; s2[t] += s2[t + s]; }
    __syncthreads();
  }
  float mu  = s1[0] * (1.0f / kD);
  float var = s2[0] * (1.0f / kD) - mu * mu;
  float inv = rsqrtf(var + 1e-5f);
  unsigned short* y = Y + (size_t)row * kD;
  y[t]       = f2bf((x0 - mu) * inv * g[t]       + bt[t]);
  y[t + 256] = f2bf((x1 - mu) * inv * g[t + 256] + bt[t + 256]);
}

// ---------------- gated RMSNorm over D=512, output bf16 ----------------
__global__ __launch_bounds__(256)
void k_grnorm(const float* __restrict__ X, const float* __restrict__ scale,
              const float* __restrict__ gate, unsigned short* __restrict__ Y)
{
  __shared__ float s2[256];
  int row = blockIdx.x, t = threadIdx.x;
  const float* x = X + (size_t)row * kD;
  float x0 = x[t], x1 = x[t + 256];
  s2[t] = x0 * x0 + x1 * x1;
  __syncthreads();
  for (int s = 128; s > 0; s >>= 1) {
    if (t < s) s2[t] += s2[t + s];
    __syncthreads();
  }
  float inv = rsqrtf(s2[0] * (1.0f / kD) + 1e-8f);
  unsigned short* y = Y + (size_t)row * kD;
  y[t]       = f2bf(x0 * inv * scale[t]       * sigmf(gate[t]       * x0));
  y[t + 256] = f2bf(x1 * inv * scale[t + 256] * sigmf(gate[t + 256] * x1));
}

// ---------------- host orchestration ----------------
extern "C" void kernel_launch(void* const* d_in, const int* in_sizes, int n_in,
                              void* d_out, int out_size, void* d_ws, size_t ws_size,
                              hipStream_t stream) {
  (void)in_sizes; (void)n_in; (void)out_size; (void)ws_size;
  const float* query   = (const float*)d_in[0];
  const float* w_in[6] = { (const float*)d_in[1],  (const float*)d_in[3],  (const float*)d_in[5],
                           (const float*)d_in[9],  (const float*)d_in[11], (const float*)d_in[13] };
  const float* b_in[6] = { (const float*)d_in[2],  (const float*)d_in[4],  (const float*)d_in[6],
                           (const float*)d_in[10], (const float*)d_in[12], (const float*)d_in[14] };
  const float* wo_lin = (const float*)d_in[7];
  const float* bo_lin = (const float*)d_in[8];
  const float* wo_loc = (const float*)d_in[15];
  const float* bo_loc = (const float*)d_in[16];
  const float* ln_g      = (const float*)d_in[17];
  const float* ln_b      = (const float*)d_in[18];
  const float* grn_scale = (const float*)d_in[19];
  const float* grn_gate  = (const float*)d_in[20];

  char* ws = (char*)d_ws;
  size_t off = 0;
  auto alloc = [&](size_t bytes) { size_t o = off; off += (bytes + 255) & ~(size_t)255; return o; };

  const size_t DE = (size_t)kD * kE;          // weight elems
  const size_t PS = (size_t)kM * kD;          // projection slot elems
  unsigned short* Xb  = (unsigned short*)(ws + alloc((size_t)kM * kE * 2));
  unsigned short* Wb  = (unsigned short*)(ws + alloc(6 * DE * 2));
  unsigned short* Wob = (unsigned short*)(ws + alloc(2 * DE * 2));
  unsigned short* P   = (unsigned short*)(ws + alloc(6 * PS * 2));
  float* o_lin = (float*)Xb;                  // kM*kD*4 bytes == kM*kE*2 bytes (reuse)
  float* o_loc = (float*)P;                   // reuses slots 0-1 (q_lin,k_lin dead by then)
  unsigned short* n1 = P + 2 * PS;            // reuses slot 2 (v_lin dead)
  unsigned short* n2 = P + 3 * PS;            // reuses slot 3 (q_loc dead)

  dim3 blk(256);
  // 1) convert query + all weights to bf16
  {
    int n = kM * kE;
    k_f2bf<<<dim3((n + 255) / 256), blk, 0, stream>>>(query, Xb, n);
    for (int i = 0; i < 6; i++)
      k_f2bf<<<dim3((int)((DE + 255) / 256)), blk, 0, stream>>>(w_in[i], Wb + i * DE, (int)DE);
    k_f2bf<<<dim3((int)((DE + 255) / 256)), blk, 0, stream>>>(wo_lin, Wob, (int)DE);
    k_f2bf<<<dim3((int)((DE + 255) / 256)), blk, 0, stream>>>(wo_loc, Wob + DE, (int)DE);
  }
  // 2) six input projections (gelu on linear-branch q,k)
  dim3 gA(kD / 128, kM / 128);
  k_gemm<1, 0><<<gA, blk, 0, stream>>>(Xb, Wb + 0 * DE, b_in[0], P + 0 * PS, kM, kD, kE);
  k_gemm<1, 0><<<gA, blk, 0, stream>>>(Xb, Wb + 1 * DE, b_in[1], P + 1 * PS, kM, kD, kE);
  k_gemm<0, 0><<<gA, blk, 0, stream>>>(Xb, Wb + 2 * DE, b_in[2], P + 2 * PS, kM, kD, kE);
  k_gemm<0, 0><<<gA, blk, 0, stream>>>(Xb, Wb + 3 * DE, b_in[3], P + 3 * PS, kM, kD, kE);
  k_gemm<0, 0><<<gA, blk, 0, stream>>>(Xb, Wb + 4 * DE, b_in[4], P + 4 * PS, kM, kD, kE);
  k_gemm<0, 0><<<gA, blk, 0, stream>>>(Xb, Wb + 5 * DE, b_in[5], P + 5 * PS, kM, kD, kE);
  // 3) linear (kernelized) branch: chunked causal recurrence
  k_attn<true><<<dim3(kB * kH), blk, 0, stream>>>(P + 0 * PS, P + 1 * PS, P + 2 * PS, o_lin);
  // 4) local branch: block-diagonal relu attention
  k_attn<false><<<dim3(kB * kH), blk, 0, stream>>>(P + 3 * PS, P + 4 * PS, P + 5 * PS, o_loc);
  // 5/6) normalizations -> bf16
  k_layernorm<<<dim3(kM), blk, 0, stream>>>(o_lin, ln_g, ln_b, n1);
  k_grnorm  <<<dim3(kM), blk, 0, stream>>>(o_loc, grn_scale, grn_gate, n2);
  // 7/8) output projections; second one averages into d_out
  dim3 gO(kE / 128, kM / 128);
  k_gemm<0, 1><<<gO, blk, 0, stream>>>(n1, Wob,      bo_lin, d_out, kM, kE, kD);
  k_gemm<0, 2><<<gO, blk, 0, stream>>>(n2, Wob + DE, bo_loc, d_out, kM, kE, kD);
}
